// CustomLoss_43834436223359
// MI455X (gfx1250) — compile-verified, matrix operand order checked by reading, not run on hardware
//
#include <hip/hip_runtime.h>

typedef __attribute__((ext_vector_type(16))) __bf16 v16bf;
typedef __attribute__((ext_vector_type(8)))  float  v8f;

#define B_Q   256
#define DIM   128
#define KNN   50
#define CHUNK 1024
#define TAUF  0.1f
#define EPSL  1e-8f
#define LAMBF 1e-4f

// LDS layout for k_knn_gemm (bytes)
#define LDS_TOPD   0
#define LDS_TOPI   51200
#define LDS_STAGE  102400
#define LDS_BSTAGE 135168          // 2 x 8192 B double-buffered TDM tile
#define LDS_TOTAL  151552

// accumulator slots in workspace
#define ACC_ANCHOR 0
#define ACC_KL     1
#define ACC_G      2
#define ACC_XX     3
#define ACC_YY     4
#define ACC_XY     5

#if __has_builtin(__builtin_amdgcn_tensor_load_to_lds) && __has_builtin(__builtin_amdgcn_s_wait_tensorcnt)
#define USE_TDM 1
#if __has_include(<hip/amd_detail/amd_gfx1250_TDM.h>)
#define TDM_ARGS6 1
#endif
#endif

__device__ __forceinline__ unsigned short f32_to_bf16u(float f) {
  union { __bf16 h; unsigned short u; } c;
  c.h = (__bf16)f;              // hardware cvt, RNE
  return c.u;
}

__device__ __forceinline__ float blk_sum256(float v, float* sred, int t) {
  sred[t] = v; __syncthreads();
  for (int s = 128; s > 0; s >>= 1) { if (t < s) sred[t] += sred[t + s]; __syncthreads(); }
  float r = sred[0]; __syncthreads();
  return r;
}

#ifdef USE_TDM
typedef __attribute__((ext_vector_type(4))) unsigned int u32x4;
typedef __attribute__((ext_vector_type(4))) int          i32x4;
typedef __attribute__((ext_vector_type(8))) int          i32x8;

// Issue one TDM load: 2D tile of 32 rows x 128 bf16 from XBf[n0..n0+31][0..127] -> LDS lds_off.
// D# per CDNA5 ISA ch.8: group0 = {count, lds_addr, global_addr, type=2}; group1 = dims/strides.
__device__ __forceinline__ void tdm_issue(const unsigned short* XBf, int n0,
                                          unsigned lds_off, int N) {
  unsigned long long ga = (unsigned long long)XBf + (unsigned long long)n0 * (DIM * 2);
  u32x4 g0;
  g0.x = 1u;                                            // count=1 (valid), user mode
  g0.y = lds_off;                                       // lds_addr (bytes)
  g0.z = (unsigned)(ga & 0xFFFFFFFFull);                // global_addr[31:0]
  g0.w = (unsigned)((ga >> 32) & 0x01FFFFFFull) | (2u << 30);   // addr[56:32] | type=2
  i32x8 g1;
  g1[0] = 0x00010000;                                   // data_size=1 (2 bytes/elem)
  g1[1] = (int)(128u << 16);                            // tensor_dim0 = 128 (lo16 @ bits 63:48)
  g1[2] = (int)(((unsigned)N & 0xFFFFu) << 16);         // tensor_dim1 lo16 @ bits 95:80
  g1[3] = (int)(((unsigned)N >> 16) | (128u << 16));    // tensor_dim1 hi16 | tile_dim0=128
  g1[4] = 32;                                           // tile_dim1 = 32 rows (tile_dim2=0)
  g1[5] = 128;                                          // tensor_dim0_stride = 128 elems
  g1[6] = 0; g1[7] = 0;
  i32x4 z4 = {0, 0, 0, 0};
#ifdef TDM_ARGS6
  i32x8 z8 = {0, 0, 0, 0, 0, 0, 0, 0};
  __builtin_amdgcn_tensor_load_to_lds(g0, g1, z4, z4, z8, 0);
#else
  __builtin_amdgcn_tensor_load_to_lds(g0, g1, z4, z4, 0);
#endif
}
#endif

// ---------------------------------------------------------------- init
__global__ void k_init(float* accum) {
  if (threadIdx.x < 16) accum[threadIdx.x] = 0.0f;
}

// ---------------------------------------------------------------- Tq = q @ W^T + b  (+ bf16 copy, norms, anchor loss)
__global__ void k_tq(const float* __restrict__ q, const float* __restrict__ W,
                     const float* __restrict__ bias, float* __restrict__ Tq,
                     unsigned short* __restrict__ TqBf, float* __restrict__ tqnorm,
                     float* accum) {
  __shared__ float qrow[DIM];
  __shared__ float red[DIM];
  int b = blockIdx.x, j = threadIdx.x;
  qrow[j] = q[b * DIM + j];
  __syncthreads();
  const float* wr = W + j * DIM;
  float s = bias[j];
  for (int k = 0; k < DIM; ++k) s = fmaf(qrow[k], wr[k], s);
  Tq[b * DIM + j]   = s;
  TqBf[b * DIM + j] = f32_to_bf16u(s);

  red[j] = s * s; __syncthreads();
  for (int st = 64; st > 0; st >>= 1) { if (j < st) red[j] += red[j + st]; __syncthreads(); }
  if (j == 0) tqnorm[b] = red[0];
  __syncthreads();
  float dq = s - qrow[j];
  red[j] = dq * dq; __syncthreads();
  for (int st = 64; st > 0; st >>= 1) { if (j < st) red[j] += red[j + st]; __syncthreads(); }
  if (j == 0) atomicAdd(&accum[ACC_ANCHOR], red[0]);
}

// ---------------------------------------------------------------- one streaming pass over X: bf16 copy + ||x||^2
__global__ void k_xcvt(const float* __restrict__ X, unsigned short* __restrict__ XBf,
                       float* __restrict__ xnorm, int N) {
  int w = (int)((blockIdx.x * blockDim.x + threadIdx.x) >> 5);
  int l = threadIdx.x & 31;
  if (w >= N) return;
  const float4* row = (const float4*)(X + (size_t)w * DIM);
  float4 v = row[l];
  float s = v.x * v.x + v.y * v.y + v.z * v.z + v.w * v.w;
  union { __bf16 h[4]; uint2 u2; } p;
  p.h[0] = (__bf16)v.x; p.h[1] = (__bf16)v.y; p.h[2] = (__bf16)v.z; p.h[3] = (__bf16)v.w;
  ((uint2*)(XBf + (size_t)w * DIM))[l] = p.u2;
  for (int off = 16; off > 0; off >>= 1) s += __shfl_xor(s, off, 32);
  if (l == 0) xnorm[w] = s;
}

// ---------------------------------------------------------------- fused bf16-WMMA distance GEMM + per-block top-50
// block = 256 threads = 8 waves; wave w owns queries [w*32, w*32+32) as two 16-row strips,
// and processes two 16-column tiles per iteration: 16 back-to-back WMMAs.
// TDM path: wave 0 DMAs the 32x128 bf16 tile into double-buffered LDS (tensor_load_to_lds +
// s_wait_tensorcnt + workgroup barrier); all 8 waves read B fragments from LDS.
__global__ void __launch_bounds__(256)
k_knn_gemm(const unsigned short* __restrict__ XBf, const unsigned short* __restrict__ TqBf,
           const float* __restrict__ tqnorm, const float* __restrict__ xnorm,
           int N, float* __restrict__ part_d, int* __restrict__ part_i) {
  extern __shared__ char smem[];
  float* topd  = (float*)(smem + LDS_TOPD);
  int*   topi  = (int*)(smem + LDS_TOPI);
  float* stage = (float*)(smem + LDS_STAGE);

  const int tid = threadIdx.x;
  const int w = tid >> 5, l = tid & 31;
  const int lane16 = l & 15;
  const int hi = l >> 4;              // 0: lanes 0-15, 1: lanes 16-31
  const int klo = hi ? 8 : 0;         // per-lane K base within each 32-K slab

  for (int i = tid; i < B_Q * KNN; i += 256) { topd[i] = 3.0e38f; topi[i] = 0; }
  __syncthreads();

  // --- preload A fragments (Tq bf16) for both strips, all 4 K-slabs ---
  union Frag { v16bf v; uint4 q2[2]; };
  Frag Af[2][4];
  float tqn[2][8];
  for (int s = 0; s < 2; ++s) {
    int m = w * 32 + s * 16 + lane16;              // A-matrix row for this lane
    const unsigned short* arow = TqBf + m * DIM;
#pragma unroll
    for (int kk = 0; kk < 4; ++kk) {
      Af[s][kk].q2[0] = *(const uint4*)(arow + kk * 32 + klo);        // K klo..klo+7
      Af[s][kk].q2[1] = *(const uint4*)(arow + kk * 32 + klo + 16);   // K klo+16..klo+23
    }
    int mbase = w * 32 + s * 16 + hi * 8;          // C-matrix rows held by this lane
    for (int r = 0; r < 8; ++r) tqn[s][r] = tqnorm[mbase + r];
  }

  const int qid = w * 32 + hi * 16 + lane16;       // query owned exclusively by this lane
  float* td = topd + qid * KNN;
  int*   ti = topi + qid * KNN;
  float wd = 3.0e38f; int wsl = 0;
  float* stw        = stage + w * 1024;
  const float* lsrd = stage + w * 1024 + hi * 256 + lane16 * 16;

  const int nb = blockIdx.x * CHUNK;
#ifdef USE_TDM
  if (w == 0) tdm_issue(XBf, nb, LDS_BSTAGE, N);   // prime buffer 0
#endif
  for (int ct = 0; ct < CHUNK / 32; ++ct) {
    int n0 = nb + ct * 32;
    int ncol0 = n0 + lane16;
    int ncol1 = n0 + 16 + lane16;
    int nc0 = ncol0 < N ? ncol0 : N - 1;
    int nc1 = ncol1 < N ? ncol1 : N - 1;

#ifdef USE_TDM
    if (w == 0) {
      if (ct + 1 < CHUNK / 32) {
        tdm_issue(XBf, n0 + 32, LDS_BSTAGE + ((unsigned)(ct + 1) & 1u) * 8192u, N);
        __builtin_amdgcn_s_wait_tensorcnt(1);      // current buffer complete
      } else {
        __builtin_amdgcn_s_wait_tensorcnt(0);
      }
    }
    __syncthreads();                               // publish buffer ct to all waves
    const unsigned short* bst =
        (const unsigned short*)(smem + LDS_BSTAGE + ((unsigned)ct & 1u) * 8192u);
    const unsigned short* brow0 = bst + lane16 * DIM;
    const unsigned short* brow1 = bst + (16 + lane16) * DIM;
#else
    const unsigned short* brow0 = XBf + (size_t)nc0 * DIM;
    const unsigned short* brow1 = XBf + (size_t)nc1 * DIM;
    { int np = ncol0 + 32; if (np >= N) np = N - 1;
      __builtin_prefetch(XBf + (size_t)np * DIM, 0, 0); }   // global_prefetch next tiles
#endif

    // ---- load ALL B fragments first (one clause burst), then the WMMA chain ----
    Frag B0[4], B1[4];
#pragma unroll
    for (int kk = 0; kk < 4; ++kk) {
      B0[kk].q2[0] = *(const uint4*)(brow0 + kk * 32 + klo);
      B0[kk].q2[1] = *(const uint4*)(brow0 + kk * 32 + klo + 16);
      B1[kk].q2[0] = *(const uint4*)(brow1 + kk * 32 + klo);
      B1[kk].q2[1] = *(const uint4*)(brow1 + kk * 32 + klo + 16);
    }

    v8f acc00 = {0.f,0.f,0.f,0.f,0.f,0.f,0.f,0.f};   // strip0 x tile0
    v8f acc01 = {0.f,0.f,0.f,0.f,0.f,0.f,0.f,0.f};   // strip0 x tile1
    v8f acc10 = {0.f,0.f,0.f,0.f,0.f,0.f,0.f,0.f};   // strip1 x tile0
    v8f acc11 = {0.f,0.f,0.f,0.f,0.f,0.f,0.f,0.f};   // strip1 x tile1
#pragma unroll
    for (int kk = 0; kk < 4; ++kk) {
      acc00 = __builtin_amdgcn_wmma_f32_16x16x32_bf16(false, Af[0][kk].v, false, B0[kk].v,
                                                      (short)0, acc00, false, false);
      acc10 = __builtin_amdgcn_wmma_f32_16x16x32_bf16(false, Af[1][kk].v, false, B0[kk].v,
                                                      (short)0, acc10, false, false);
      acc01 = __builtin_amdgcn_wmma_f32_16x16x32_bf16(false, Af[0][kk].v, false, B1[kk].v,
                                                      (short)0, acc01, false, false);
      acc11 = __builtin_amdgcn_wmma_f32_16x16x32_bf16(false, Af[1][kk].v, false, B1[kk].v,
                                                      (short)0, acc11, false, false);
    }

    float xnv0 = xnorm[nc0];
    float xnv1 = xnorm[nc1];
#pragma unroll
    for (int r = 0; r < 8; ++r) {
      int ro = (hi * 8 + r) * 16 + lane16;
      float d;
      d = tqn[0][r] + xnv0 - 2.0f * acc00[r]; d = d > 0.f ? d : 0.f; stw[ro]              = d;
      d = tqn[1][r] + xnv0 - 2.0f * acc10[r]; d = d > 0.f ? d : 0.f; stw[256 + ro]        = d;
      d = tqn[0][r] + xnv1 - 2.0f * acc01[r]; d = d > 0.f ? d : 0.f; stw[512 + ro]        = d;
      d = tqn[1][r] + xnv1 - 2.0f * acc11[r]; d = d > 0.f ? d : 0.f; stw[512 + 256 + ro]  = d;
    }
    // LDS is in-order within a wave: safe to read other lanes' stage data.
    for (int tt = 0; tt < 2; ++tt) {
      int nt = n0 + tt * 16;
      if (nt >= N) break;                           // uniform across lanes
      const float* ls = lsrd + tt * 512;
      for (int c = 0; c < 16; ++c) {
        int n = nt + c;
        if (n >= N) break;                          // uniform across lanes
        float d = ls[c];
        if (d < wd) {
          td[wsl] = d; ti[wsl] = n;
          float mx = td[0]; int ms = 0;
          for (int j = 1; j < KNN; ++j) { float v = td[j]; if (v > mx) { mx = v; ms = j; } }
          wd = mx; wsl = ms;
        }
      }
    }
#ifdef USE_TDM
    __syncthreads();                               // all reads done before buffer reuse
#endif
  }

  size_t obase = ((size_t)blockIdx.x * B_Q + qid) * KNN;
  for (int j = 0; j < KNN; ++j) { part_d[obase + j] = td[j]; part_i[obase + j] = ti[j]; }
}

// ---------------------------------------------------------------- merge partial top-50 -> exact l2 -> softmax -> union-KL
__global__ void k_merge(const float* __restrict__ part_d, const int* __restrict__ part_i,
                        int nblk, const float* __restrict__ Tq, const float* __restrict__ X,
                        const float* __restrict__ pre_weights, const int* __restrict__ pre_indices,
                        const int* __restrict__ q_indices, float* accum) {
  __shared__ float rv[256]; __shared__ int ri[256];
  __shared__ float sred[256];
  __shared__ int   sel_i[KNN];
  __shared__ float l2s[KNN], post_w[KNN], pre_wS[KNN];
  __shared__ int   pre_iS[KNN];
  __shared__ int   cat[2 * KNN];

  int qrow = blockIdx.x, t = threadIdx.x;
  int tot = nblk * KNN;

  float cd[40]; int ci[40]; bool used[40];
  int cnt = 0;
  for (int i = t; i < tot && cnt < 40; i += 256) {
    int blk = i / KNN, k = i % KNN;
    size_t off = ((size_t)blk * B_Q + qrow) * KNN + k;
    cd[cnt] = part_d[off]; ci[cnt] = part_i[off]; used[cnt] = false; ++cnt;
  }

  for (int r = 0; r < KNN; ++r) {
    float best = 3.0e38f; int bslot = -1;
    for (int i = 0; i < cnt; ++i)
      if (!used[i] && cd[i] < best) { best = cd[i]; bslot = i; }
    rv[t] = best; ri[t] = t; __syncthreads();
    for (int s = 128; s > 0; s >>= 1) {
      if (t < s) { if (rv[t + s] < rv[t]) { rv[t] = rv[t + s]; ri[t] = ri[t + s]; } }
      __syncthreads();
    }
    if (t == ri[0] && bslot >= 0) { used[bslot] = true; sel_i[r] = ci[bslot]; }
    __syncthreads();
  }

  if (t < KNN) {                    // exact f32 l2 to selected neighbors (grad path in ref)
    const float* tq = Tq + qrow * DIM;
    const float* xr = X + (size_t)sel_i[t] * DIM;
    float s = 0.f;
    for (int k = 0; k < DIM; ++k) { float df = tq[k] - xr[k]; s = fmaf(df, df, s); }
    l2s[t] = s;
  }
  __syncthreads();
  if (t == 0) {                     // softmax(-l2/tau) over K=50
    float mx = -3.0e38f;
    for (int k = 0; k < KNN; ++k) { float v = -l2s[k] / TAUF; if (v > mx) mx = v; }
    float den = 0.f;
    for (int k = 0; k < KNN; ++k) den += expf(-l2s[k] / TAUF - mx);
    for (int k = 0; k < KNN; ++k) post_w[k] = expf(-l2s[k] / TAUF - mx) / den;
  }
  if (t < KNN) {
    int qi = q_indices[qrow];
    pre_iS[t] = pre_indices[qi * KNN + t];
    pre_wS[t] = pre_weights[qi * KNN + t];
  }
  __syncthreads();
  if (t < 2 * KNN) cat[t] = (t < KNN) ? pre_iS[t] : sel_i[t - KNN];
  __syncthreads();

  float spv = 0.f, sqv = 0.f, pcv = 0.f, qcv = 0.f, multv = 1.f;
  if (t < 2 * KNN) {
    int id = cat[t];
    float m = 0.f;
    for (int k = 0; k < 2 * KNN; ++k) m += (cat[k] == id) ? 1.f : 0.f;
    float pr = 0.f, qr = 0.f;
    for (int k = 0; k < KNN; ++k) {
      pr += (pre_iS[k] == id) ? pre_wS[k] : 0.f;
      qr += (sel_i[k] == id) ? post_w[k] : 0.f;
    }
    pcv = pr > EPSL ? pr : EPSL;
    qcv = qr > EPSL ? qr : EPSL;
    multv = m;
    spv = pcv / m; sqv = qcv / m;
  }
  float Sp = blk_sum256(spv, sred, t);
  float Sq = blk_sum256(sqv, sred, t);
  float term = 0.f;
  if (t < 2 * KNN) {
    float p = pcv / Sp, qq = qcv / Sq;
    term = (p * (logf(p) - logf(qq))) / multv;
  }
  float kl = blk_sum256(term, sred, t);
  if (t == 0) atomicAdd(&accum[ACC_KL], kl);
}

// ---------------------------------------------------------------- comb norms & 512x512 d2 for MMD
__global__ void k_combnorm(const float* __restrict__ tqnorm, const float* __restrict__ xnorm,
                           const int* __restrict__ idx, float* __restrict__ cn) {
  int i = blockIdx.x * blockDim.x + threadIdx.x;
  if (i < 512) cn[i] = (i < 256) ? tqnorm[i] : xnorm[idx[i - 256]];
}

__global__ void k_combd2(const float* __restrict__ Tq, const float* __restrict__ X,
                         const int* __restrict__ idx, const float* __restrict__ cn,
                         float* __restrict__ d2) {
  __shared__ float rowi[DIM];
  int i = blockIdx.x, t = threadIdx.x;
  const float* ri = (i < 256) ? (Tq + i * DIM) : (X + (size_t)idx[i - 256] * DIM);
  if (t < DIM) rowi[t] = ri[t];
  __syncthreads();
  float ni = cn[i];
  for (int j = t; j < 512; j += blockDim.x) {
    const float* rj = (j < 256) ? (Tq + j * DIM) : (X + (size_t)idx[j - 256] * DIM);
    float dot = 0.f;
    for (int k = 0; k < DIM; ++k) dot = fmaf(rowi[k], rj[k], dot);
    float v = ni + cn[j] - 2.f * dot;
    d2[i * 512 + j] = v > 0.f ? v : 0.f;
  }
}

// ---------------------------------------------------------------- exact median (2 rank-selects) -> g
__global__ void k_median(const float* __restrict__ d2, float* accum) {
  __shared__ float red[1024];
  __shared__ int  ired[1024];
  int t = threadIdx.x;
  const int TOT = 512 * 512;
  float mx = 0.f;
  for (int i = t; i < TOT; i += 1024) { float v = d2[i]; mx = v > mx ? v : mx; }
  red[t] = mx; __syncthreads();
  for (int s = 512; s > 0; s >>= 1) { if (t < s) red[t] = red[t] > red[t + s] ? red[t] : red[t + s]; __syncthreads(); }
  float maxv = red[0]; __syncthreads();

  float med[2];
  int ranks[2] = { TOT / 2, TOT / 2 + 1 };   // 1-based middle order statistics
  for (int r = 0; r < 2; ++r) {
    float lo = 0.f, hi = maxv;
    for (int it = 0; it < 48; ++it) {
      float mid = 0.5f * (lo + hi);
      int c = 0;
      for (int i = t; i < TOT; i += 1024) c += (d2[i] <= mid) ? 1 : 0;
      ired[t] = c; __syncthreads();
      for (int s = 512; s > 0; s >>= 1) { if (t < s) ired[t] += ired[t + s]; __syncthreads(); }
      int cntv = ired[0]; __syncthreads();
      if (cntv >= ranks[r]) hi = mid; else lo = mid;
    }
    med[r] = hi;
  }
  if (t == 0) {
    float median = 0.5f * (med[0] + med[1]);
    float sig = median * 0.5f;
    if (sig < 1e-6f) sig = 1.0f;
    accum[ACC_G] = 1.0f / (sig + EPSL);
  }
}

// ---------------------------------------------------------------- kxx / kyy / kxy block sums
__global__ void k_ksum(const float* __restrict__ d2, float* accum) {
  __shared__ float sred[256];
  int t = threadIdx.x;
  float g = accum[ACC_G];
  float xx = 0.f, yy = 0.f, xy = 0.f;
  for (int e = blockIdx.x * 256 + t; e < 512 * 512; e += gridDim.x * 256) {
    int i = e >> 9, j = e & 511;
    float v = expf(-g * d2[e]);
    if (i < 256) { if (j < 256) xx += v; else xy += v; }
    else if (j >= 256) yy += v;
  }
  float s;
  s = blk_sum256(xx, sred, t); if (t == 0) atomicAdd(&accum[ACC_XX], s);
  s = blk_sum256(yy, sred, t); if (t == 0) atomicAdd(&accum[ACC_YY], s);
  s = blk_sum256(xy, sred, t); if (t == 0) atomicAdd(&accum[ACC_XY], s);
}

// ---------------------------------------------------------------- finalize
__global__ void k_final(const float* __restrict__ W, const float* __restrict__ bias,
                        const float* __restrict__ accum, float* __restrict__ out) {
  __shared__ float red[256];
  int t = threadIdx.x;
  float s = 0.f;
  for (int i = t; i < DIM * DIM; i += 256) s += W[i] * W[i];
  if (t < DIM) s += bias[t] * bias[t];
  red[t] = s; __syncthreads();
  for (int st = 128; st > 0; st >>= 1) { if (t < st) red[t] += red[t + st]; __syncthreads(); }
  if (t == 0) {
    float reg = 0.5f * red[0];
    float kxx = accum[ACC_XX] / 65536.f;
    float kyy = accum[ACC_YY] / 65536.f;
    float kxy = accum[ACC_XY] / 65536.f;
    float ld = kxx + kyy - 2.f * kxy; ld = ld > 0.f ? ld : 0.f;
    float lk = accum[ACC_KL] / 256.f;
    float la = accum[ACC_ANCHOR] / 256.f;
    float total = ld + lk + LAMBF * reg + la;
    out[0] = total; out[1] = ld; out[2] = lk; out[3] = la;
  }
}

// ---------------------------------------------------------------- host
extern "C" void kernel_launch(void* const* d_in, const int* in_sizes, int n_in,
                              void* d_out, int out_size, void* d_ws, size_t ws_size,
                              hipStream_t stream) {
  const float* q_batch     = (const float*)d_in[0];
  const float* X           = (const float*)d_in[1];
  const float* W           = (const float*)d_in[2];
  const float* bvec        = (const float*)d_in[3];
  const float* pre_weights = (const float*)d_in[4];
  const int*   pre_indices = (const int*)d_in[5];
  const int*   q_indices   = (const int*)d_in[6];
  const int*   idx         = (const int*)d_in[7];

  const int N    = in_sizes[1] / DIM;
  const int nblk = (N + CHUNK - 1) / CHUNK;

  char* ws = (char*)d_ws;
  size_t off = 0;
  auto alloc = [&](size_t bytes) { size_t p = off; off += (bytes + 255) & ~(size_t)255; return p; };
  float*          Tq       = (float*)(ws + alloc((size_t)B_Q * DIM * 4));
  unsigned short* TqBf     = (unsigned short*)(ws + alloc((size_t)B_Q * DIM * 2));
  float*          tqnorm   = (float*)(ws + alloc(B_Q * 4));
  float*          combnorm = (float*)(ws + alloc(512 * 4));
  float*          accum    = (float*)(ws + alloc(64 * 4));
  float*          xnorm    = (float*)(ws + alloc((size_t)N * 4));
  unsigned short* XBf      = (unsigned short*)(ws + alloc((size_t)N * DIM * 2));
  float*          d2c      = (float*)(ws + alloc((size_t)512 * 512 * 4));
  float*          part_d   = (float*)(ws + alloc((size_t)nblk * B_Q * KNN * 4));
  int*            part_i   = (int*)(ws + alloc((size_t)nblk * B_Q * KNN * 4));
  (void)n_in; (void)out_size; (void)ws_size;

  k_init<<<1, 32, 0, stream>>>(accum);
  k_tq<<<B_Q, DIM, 0, stream>>>(q_batch, W, bvec, Tq, TqBf, tqnorm, accum);
  k_xcvt<<<(N + 7) / 8, 256, 0, stream>>>(X, XBf, xnorm, N);

  k_knn_gemm<<<nblk, 256, LDS_TOTAL, stream>>>(XBf, TqBf, tqnorm, xnorm, N, part_d, part_i);

  k_merge<<<B_Q, 256, 0, stream>>>(part_d, part_i, nblk, Tq, X,
                                   pre_weights, pre_indices, q_indices, accum);
  k_combnorm<<<2, 256, 0, stream>>>(tqnorm, xnorm, idx, combnorm);
  k_combd2<<<512, 256, 0, stream>>>(Tq, X, idx, combnorm, d2c);
  k_median<<<1, 1024, 0, stream>>>(d2c, accum);
  k_ksum<<<256, 256, 0, stream>>>(d2c, accum);
  k_final<<<1, 256, 0, stream>>>(W, bvec, accum, (float*)d_out);
}
